// GraphAttention_764504178943
// MI455X (gfx1250) — matrix operand order repaired and checked by reading, not compiled, not run on hardware
//
#include <hip/hip_runtime.h>
#include <hip/hip_bf16.h>
#include <cstdint>

// ---- problem constants ------------------------------------------------------
#define BSZ     4
#define NNODE   2048
#define NINP    128
#define NFEAT   64
#define NHEADS  4
#define NCOL    (NHEADS * NFEAT)          // 256 concatenated projection cols
#define MROWS   (BSZ * NNODE)             // 8192 projection rows

typedef __attribute__((ext_vector_type(16))) _Float16 v16h;
typedef __attribute__((ext_vector_type(8)))  _Float16 v8h;
typedef __attribute__((ext_vector_type(4)))  _Float16 v4h;
typedef __attribute__((ext_vector_type(8)))  float    v8f;
typedef __attribute__((ext_vector_type(4)))  float    v4f;

// ---- workspace layout (bytes) ----------------------------------------------
#define OFF_INP16  0
#define OFF_WT16   (2097152)
#define OFF_H32    (OFF_WT16 + 65536)
#define OFF_HT16   (OFF_H32 + 8388608)
#define OFF_SSRC   (OFF_HT16 + 4194304)
#define OFF_SDST   (OFF_SSRC + 131072)

// =============================================================================
// Kernel 0: f32 -> f16 conversion of inp (vectorized 4-wide), W -> WT[col][k]
// =============================================================================
__global__ void gat_cvt_kernel(const float* __restrict__ inp,
                               const float* __restrict__ W,
                               _Float16* __restrict__ inp16,
                               _Float16* __restrict__ wt16) {
  int idx = blockIdx.x * blockDim.x + threadIdx.x;
  const int n_inp4 = (MROWS * NINP) / 4;          // 262,144 float4 chunks
  if (idx < n_inp4) {
    v4f x = *(const v4f*)(inp + idx * 4);
    v4h y;
    #pragma unroll
    for (int e = 0; e < 4; ++e) y[e] = (_Float16)x[e];
    *(v4h*)(inp16 + idx * 4) = y;
  } else {
    int c = idx - n_inp4;                         // 0 .. NCOL*NINP-1
    if (c < NCOL * NINP) {
      int col = c >> 7;                           // 0..255  (= head*64 + f)
      int k   = c & 127;                          // 0..127
      int head = col >> 6, f = col & 63;
      wt16[col * NINP + k] =
          (_Float16)W[((head * NINP) + k) * NFEAT + f];
    }
  }
}

// =============================================================================
// Kernel 1: projection GEMM  h = inp @ Wcat + bias  via v_wmma_f32_16x16x32_f16
// One wave per 16x16 output tile. grid = (MROWS/16, NCOL/16).
// =============================================================================
__global__ __launch_bounds__(32)
void gat_proj_kernel(const _Float16* __restrict__ inp16,
                     const _Float16* __restrict__ wt16,
                     const float* __restrict__ bias,     // (H,F) == flat 256
                     float* __restrict__ h32,
                     _Float16* __restrict__ ht16) {
  const int tile_m = blockIdx.x;          // 0..511
  const int tile_n = blockIdx.y;          // 0..15
  const int l      = threadIdx.x;         // 0..31
  const int lo16   = (l < 16);

  const int arow = tile_m * 16 + (l & 15);
  const _Float16* aptr = inp16 + arow * NINP + (lo16 ? 0 : 16);
  const int col  = tile_n * 16 + (l & 15);
  const _Float16* bptr = wt16 + col * NINP + (lo16 ? 0 : 16);

  v8f acc = {};
  #pragma unroll
  for (int k = 0; k < NINP; k += 32) {
    v8h alo = *(const v8h*)(aptr + k);        // K = kb16 + 0..7
    v8h ahi = *(const v8h*)(aptr + k + 8);    // K = kb16 + 8..15
    v16h a;
    #pragma unroll
    for (int e = 0; e < 8; ++e) { a[e] = alo[e]; a[8 + e] = ahi[e]; }
    v16h b = *(const v16h*)(bptr + k);        // K = kb16 + 0..15
    acc = __builtin_amdgcn_wmma_f32_16x16x32_f16(
        false, a, false, b, (short)0, acc, false, false);
  }

  const float bv = bias[col];
  const int head = col >> 6, f = col & 63;
  const int rbase = tile_m * 16 + (lo16 ? 0 : 8);
  const int bb   = rbase >> 11;
  const int n0   = rbase & 2047;

  v8h hv;
  #pragma unroll
  for (int v = 0; v < 8; ++v) {
    float val = acc[v] + bv;
    h32[(((size_t)(bb * NHEADS + head) * NNODE) + (n0 + v)) * NFEAT + f] = val;
    hv[v] = (_Float16)val;
  }
  *(v8h*)(ht16 + ((size_t)(bb * NHEADS + head) * NFEAT + f) * NNODE + n0) = hv;
}

// =============================================================================
// Kernel 2: per-node score dots  ssrc = h.a_src + a_bias, sdst = h.a_dst
// =============================================================================
__global__ void gat_sdots_kernel(const float* __restrict__ h32,
                                 const float* __restrict__ a_src,
                                 const float* __restrict__ a_dst,
                                 const float* __restrict__ a_bias,
                                 float* __restrict__ ssrc,
                                 float* __restrict__ sdst) {
  int tid = blockIdx.x * blockDim.x + threadIdx.x;     // (b*H+h)*N + n
  if (tid >= BSZ * NHEADS * NNODE) return;
  int head = (tid >> 11) & (NHEADS - 1);
  const float* hp = h32 + (size_t)tid * NFEAT;
  const float* as = a_src + head * NFEAT;
  const float* ad = a_dst + head * NFEAT;
  float si = 0.f, sj = 0.f;
  #pragma unroll
  for (int f = 0; f < NFEAT; ++f) { si += hp[f] * as[f]; sj += hp[f] * ad[f]; }
  ssrc[tid] = si + a_bias[head];
  sdst[tid] = sj;
}

// =============================================================================
// Kernel 3: fused masked softmax + NT att write + WMMA aggregation + ELU
// grid = (NNODE/16 row tiles, B*H), block = 128 (4 waves).
// =============================================================================
#define PSTRIDE 2064                               // halves; 16B-aligned rows
#define SMEM_BYTES (16 * PSTRIDE * 2 + (128 + 16 + 16 + 16) * 4)

__global__ __launch_bounds__(128)
void gat_attn_kernel(const unsigned char* __restrict__ adj,
                     const float* __restrict__ ssrc,
                     const float* __restrict__ sdst,
                     const _Float16* __restrict__ ht16,
                     float* __restrict__ out,
                     float* __restrict__ att) {
  extern __shared__ char smem[];
  _Float16* pbuf  = (_Float16*)smem;                       // 16 x PSTRIDE
  float* red      = (float*)(smem + 16 * PSTRIDE * 2);     // [16][8]
  float* rowmax   = red + 128;
  float* rowinv   = rowmax + 16;
  float* srow     = rowinv + 16;

  const int bh = blockIdx.y;                 // b*NHEADS + h
  const int i0 = blockIdx.x * 16;
  const int t  = threadIdx.x;
  const float* sj = sdst + (size_t)bh * NNODE;

  if (t < 16) srow[t] = ssrc[(size_t)bh * NNODE + i0 + t];
  __syncthreads();

  const int r = t >> 3, sub = t & 7;
  const unsigned char* arow = adj + (size_t)(i0 + r) * NNODE;
  const float s_ir = srow[r];

  // ---- pass 1: row max, 4 j per step (u32 adj load + float4 sj load)
  float m = -1e9f;
  for (int j = sub * 4; j < NNODE; j += 32) {
    unsigned am = *(const unsigned*)(arow + j);
    if (am) {
      v4f sv = *(const v4f*)(sj + j);
      if (am & 0x000000ffu) m = fmaxf(m, s_ir + sv[0]);
      if (am & 0x0000ff00u) m = fmaxf(m, s_ir + sv[1]);
      if (am & 0x00ff0000u) m = fmaxf(m, s_ir + sv[2]);
      if (am & 0xff000000u) m = fmaxf(m, s_ir + sv[3]);
    }
  }
  red[r * 8 + sub] = m;
  __syncthreads();
  if (sub == 0) {
    float mm = red[r * 8];
    #pragma unroll
    for (int q = 1; q < 8; ++q) mm = fmaxf(mm, red[r * 8 + q]);
    rowmax[r] = mm;
  }
  __syncthreads();

  // ---- pass 2: exp-sum, 4 j per step
  const float rm = rowmax[r];
  float s = 0.f;
  for (int j = sub * 4; j < NNODE; j += 32) {
    unsigned am = *(const unsigned*)(arow + j);
    if (am) {
      v4f sv = *(const v4f*)(sj + j);
      if (am & 0x000000ffu) s += __expf(s_ir + sv[0] - rm);
      if (am & 0x0000ff00u) s += __expf(s_ir + sv[1] - rm);
      if (am & 0x00ff0000u) s += __expf(s_ir + sv[2] - rm);
      if (am & 0xff000000u) s += __expf(s_ir + sv[3] - rm);
    }
  }
  red[r * 8 + sub] = s;
  __syncthreads();
  if (sub == 0) {
    float ss = 0.f;
    #pragma unroll
    for (int q = 0; q < 8; ++q) ss += red[r * 8 + q];
    rowinv[r] = 1.0f / ss;
  }
  __syncthreads();

  // ---- pass 3: normalized p -> att (NT b128 stores, 2KB per row sweep)
  //              and LDS f16 p-buffer (b64 stores)
  float* attb = att + ((size_t)bh * NNODE + i0) * NNODE;
  for (int rr = 0; rr < 16; ++rr) {
    const float rmm = rowmax[rr], rin = rowinv[rr], sirr = srow[rr];
    const unsigned char* ar = adj + (size_t)(i0 + rr) * NNODE;
    #pragma unroll
    for (int k = 0; k < NNODE / 512; ++k) {
      int j = t * 4 + k * 512;
      unsigned am = *(const unsigned*)(ar + j);
      v4f sv = *(const v4f*)(sj + j);
      v4f pv;
      pv[0] = (am & 0x000000ffu) ? __expf(sirr + sv[0] - rmm) * rin : 0.f;
      pv[1] = (am & 0x0000ff00u) ? __expf(sirr + sv[1] - rmm) * rin : 0.f;
      pv[2] = (am & 0x00ff0000u) ? __expf(sirr + sv[2] - rmm) * rin : 0.f;
      pv[3] = (am & 0xff000000u) ? __expf(sirr + sv[3] - rmm) * rin : 0.f;
      __builtin_nontemporal_store(pv, (v4f*)(attb + (size_t)rr * NNODE + j));
      v4h ph;
      #pragma unroll
      for (int e = 0; e < 4; ++e) ph[e] = (_Float16)pv[e];
      *(v4h*)(pbuf + rr * PSTRIDE + j) = ph;
    }
  }
  __syncthreads();

  // ---- pass 4: out_tile[16x64] = P[16x2048] @ H[2048x64] via f16 WMMA
  const int w    = t >> 5;                   // wave id -> 16-col slab of F
  const int lane = t & 31;
  const int lo16 = (lane < 16);
  const int f    = w * 16 + (lane & 15);
  const _Float16* hptr =
      ht16 + ((size_t)bh * NFEAT + f) * NNODE + (lo16 ? 0 : 16);
  const _Float16* pp = pbuf + (lane & 15) * PSTRIDE + (lo16 ? 0 : 16);

  v8f acc = {};
  for (int jc = 0; jc < NNODE; jc += 32) {
    v8h alo = *(const v8h*)(pp + jc);        // K = kb16 + 0..7
    v8h ahi = *(const v8h*)(pp + jc + 8);    // K = kb16 + 8..15
    v16h a;
    #pragma unroll
    for (int e = 0; e < 8; ++e) { a[e] = alo[e]; a[8 + e] = ahi[e]; }
    v16h b = *(const v16h*)(hptr + jc);      // K contiguous, column = f
    acc = __builtin_amdgcn_wmma_f32_16x16x32_f16(
        false, a, false, b, (short)0, acc, false, false);
  }

  // ---- pass 5: ELU + NT store out (B, N, H*F)
  const int bb = bh >> 2, head = bh & 3;
  const int rbase = i0 + (lo16 ? 0 : 8);
  #pragma unroll
  for (int v = 0; v < 8; ++v) {
    float x = acc[v];
    float e = (x > 0.f) ? x : (__expf(x) - 1.f);
    __builtin_nontemporal_store(
        e, out + ((size_t)bb * NNODE + (rbase + v)) * NCOL + head * NFEAT + f);
  }
}

// =============================================================================
extern "C" void kernel_launch(void* const* d_in, const int* in_sizes, int n_in,
                              void* d_out, int out_size, void* d_ws, size_t ws_size,
                              hipStream_t stream) {
  const float*         inp    = (const float*)d_in[0];
  const unsigned char* adj    = (const unsigned char*)d_in[1];   // jnp bool
  const float*         W      = (const float*)d_in[2];
  const float*         bias   = (const float*)d_in[3];
  const float*         a_src  = (const float*)d_in[4];
  const float*         a_dst  = (const float*)d_in[5];
  const float*         a_bias = (const float*)d_in[6];

  float* out = (float*)d_out;                               // (B,N,256)
  float* att = out + (size_t)BSZ * NNODE * NCOL;            // (B,H,N,N)

  char* ws = (char*)d_ws;
  _Float16* inp16 = (_Float16*)(ws + OFF_INP16);
  _Float16* wt16  = (_Float16*)(ws + OFF_WT16);
  float*    h32   = (float*)   (ws + OFF_H32);
  _Float16* ht16  = (_Float16*)(ws + OFF_HT16);
  float*    ssrc  = (float*)   (ws + OFF_SSRC);
  float*    sdst  = (float*)   (ws + OFF_SDST);

  // 0) convert inp + W to f16
  {
    int total = (MROWS * NINP) / 4 + NCOL * NINP;
    gat_cvt_kernel<<<(total + 255) / 256, 256, 0, stream>>>(inp, W, inp16, wt16);
  }
  // 1) projection GEMM (WMMA f16)
  gat_proj_kernel<<<dim3(MROWS / 16, NCOL / 16), 32, 0, stream>>>(
      inp16, wt16, bias, h32, ht16);
  // 2) score dot products
  gat_sdots_kernel<<<(BSZ * NHEADS * NNODE) / 256, 256, 0, stream>>>(
      h32, a_src, a_dst, a_bias, ssrc, sdst);
  // 3) fused softmax + att + aggregation (WMMA f16) + ELU
  gat_attn_kernel<<<dim3(NNODE / 16, BSZ * NHEADS), 128, SMEM_BYTES, stream>>>(
      adj, ssrc, sdst, ht16, out, att);
}
